// Net_83339545412194
// MI455X (gfx1250) — matrix operand order repaired and checked by reading, not compiled, not run on hardware
//
#include <hip/hip_runtime.h>
#include <hip/hip_bf16.h>

// ---------------------------------------------------------------------------
// CDNA5 / gfx1250 QA-reader forward pass.
// All GEMMs: v_wmma_f32_16x16x32_f16 with f32 accumulate.
// All WMMA operands are pre-converted to f16 in padded buffers so fragment
// fills are unconditional 16B-aligned vector loads (global_load_b128 /
// ds_load_b128) -- no per-element guards, no serialized waits.
// BiLSTM recurrences: persistent single-workgroup kernels, h/c in LDS,
// per-step h@Wh^T as WMMA tiles.
// ---------------------------------------------------------------------------

typedef __attribute__((ext_vector_type(16))) _Float16 v16h;
typedef __attribute__((ext_vector_type(8)))  _Float16 h8;
typedef __attribute__((ext_vector_type(8)))  float    v8f;

#define Nn   32
#define Ll   400
#define Mm   30
#define Ee   300
#define Hh   100
#define H4   400   // 4*H gate width
#define Aa   30
#define KPAD 128   // H padded for WMMA K-chunks

union fragU { v16h v; h8 h[2]; };

// A-frag (16x32 f16): lane holds row lane&15; halves k0+hi8+[0..7] and +16.
__device__ __forceinline__ v16h load_fragA(const _Float16* rowk0, int lane) {
  const _Float16* p = rowk0 + ((lane >> 4) << 3);
  fragU u;
  u.h[0] = *(const h8*)(p);
  u.h[1] = *(const h8*)(p + 16);
  return u.v;
}
// B-frag (32x16 f16): lane holds col lane&15; halves k0+hi8+[0..7] and +16.
__device__ __forceinline__ v16h load_fragB(const _Float16* colk0, int lane) {
  const _Float16* p = colk0 + ((lane >> 4) << 3);
  fragU u;
  u.h[0] = *(const h8*)(p);
  u.h[1] = *(const h8*)(p + 16);
  return u.v;
}

__device__ __forceinline__ v8f wmma_acc(v16h a, v16h b, v8f c) {
  return __builtin_amdgcn_wmma_f32_16x16x32_f16(false, a, false, b, (short)0,
                                                c, false, false);
}

// ---------------------------------------------------------------------------
// Conversion / gather / pack kernels
// ---------------------------------------------------------------------------

__global__ void k_gather(const float* __restrict__ emdW,
                         const int* __restrict__ tok,
                         float* __restrict__ out, int R) {
  int idx = blockIdx.x * blockDim.x + threadIdx.x;
  if (idx >= R * Ee) return;
  int r = idx / Ee, e = idx % Ee;
  out[idx] = emdW[(size_t)tok[r] * Ee + e];
}

// Batched f32 -> f16 with row/col zero padding.
__global__ void k_h_pad(const float* __restrict__ src, _Float16* __restrict__ dst,
                        int Bc, int Rs, int Rp, int Ks, int Kp) {
  int idx = blockIdx.x * blockDim.x + threadIdx.x;
  if (idx >= Bc * Rp * Kp) return;
  int k = idx % Kp;
  int r = (idx / Kp) % Rp;
  int b = idx / (Kp * Rp);
  dst[idx] = (r < Rs && k < Ks)
                 ? (_Float16)src[((size_t)b * Rs + r) * Ks + k]
                 : (_Float16)0.f;
}

// Wh (400x100 fp32) -> f16 [col j][k] padded to KPAD.
__global__ void k_wh2half(const float* __restrict__ Wh,
                          _Float16* __restrict__ out) {
  int idx = blockIdx.x * blockDim.x + threadIdx.x;
  if (idx >= H4 * KPAD) return;
  int j = idx / KPAD, k = idx % KPAD;
  out[idx] = (k < Hh) ? (_Float16)Wh[j * Hh + k] : (_Float16)0.f;
}

// q_emd transpose: qeT[n][e(304)][m(32)] = q_emd[n][m][e]  (f16, padded)
__global__ void k_qeT(const float* __restrict__ qe, _Float16* __restrict__ out) {
  int idx = blockIdx.x * blockDim.x + threadIdx.x;
  if (idx >= Nn * 304 * 32) return;
  int m = idx % 32;
  int e = (idx / 32) % 304;
  int n = idx / (32 * 304);
  out[idx] = (e < Ee && m < Mm)
                 ? (_Float16)qe[((size_t)n * Mm + m) * Ee + e]
                 : (_Float16)0.f;
}

// concat f/b scan outputs (t-major) -> f16 t-major rows of width 224 (padded)
__global__ void k_concat_h_t(const float* __restrict__ yf,
                             const float* __restrict__ yb,
                             _Float16* __restrict__ out, int T) {
  int idx = blockIdx.x * blockDim.x + threadIdx.x;
  if (idx >= T * Nn * 224) return;
  int k = idx % 224;
  int r = idx / 224;
  float v = 0.f;
  if (k < Hh)           v = yf[(size_t)r * Hh + k];
  else if (k < 2 * Hh)  v = yb[(size_t)r * Hh + (k - Hh)];
  out[idx] = (_Float16)v;
}

// concat f/b scan outputs (t-major) -> f16 n-major (n,t,224 padded)
__global__ void k_concat_h_n(const float* __restrict__ yf,
                             const float* __restrict__ yb,
                             _Float16* __restrict__ out, int T) {
  int idx = blockIdx.x * blockDim.x + threadIdx.x;
  if (idx >= Nn * T * 224) return;
  int k = idx % 224;
  int t = (idx / 224) % T;
  int n = idx / (224 * T);
  float v = 0.f;
  if (k < Hh)          v = yf[((size_t)t * Nn + n) * Hh + k];
  else if (k < 2 * Hh) v = yb[((size_t)t * Nn + n) * Hh + (k - Hh)];
  out[idx] = (_Float16)v;
}

// concat f/b scan outputs (t-major) -> f32 n-major (n,t,2H)  (for q_indep)
__global__ void k_concat(const float* __restrict__ yf,
                         const float* __restrict__ yb,
                         float* __restrict__ out, int T) {
  int idx = blockIdx.x * blockDim.x + threadIdx.x;
  if (idx >= Nn * T * 2 * Hh) return;
  int k = idx % (2 * Hh);
  int t = (idx / (2 * Hh)) % T;
  int n = idx / (2 * Hh * T);
  out[idx] = (k < Hh) ? yf[((size_t)t * Nn + n) * Hh + k]
                      : yb[((size_t)t * Nn + n) * Hh + (k - Hh)];
}

// p_star assembled directly in f16 (only the gate GEMM consumes it)
__global__ void k_pstar_h(const float* __restrict__ pe,
                          const float* __restrict__ qal,
                          const float* __restrict__ qind,
                          _Float16* __restrict__ out) {
  int idx = blockIdx.x * blockDim.x + threadIdx.x;
  if (idx >= Nn * Ll * 800) return;
  int k = idx % 800;
  int l = (idx / 800) % Ll;
  int n = idx / (800 * Ll);
  float v;
  if (k < Ee)          v = pe[((size_t)n * Ll + l) * Ee + k];
  else if (k < 2 * Ee) v = qal[((size_t)n * Ll + l) * Ee + (k - Ee)];
  else                 v = qind[n * 2 * Hh + (k - 2 * Ee)];
  out[idx] = (_Float16)v;
}

// ---------------------------------------------------------------------------
// WMMA GEMM: out[r,j] = relu(sum_k X[r,k]*W[j,k] + bias[j]) * mask(r)
// X: f16 (R x Kp), rows n-major r = n*T + t.  W: f16 (Houtp x Kp).
// R multiple of 16; Kp multiple of 32; Houtp = ceil16(Hout).
// ---------------------------------------------------------------------------
__global__ void k_ff(const _Float16* __restrict__ X, const _Float16* __restrict__ W,
                     const float* __restrict__ bias,
                     const int* __restrict__ lens, float* __restrict__ out,
                     int R, int Kp, int Hout, int T) {
  const int lane = threadIdx.x & 31;
  const int wid = (blockIdx.x * blockDim.x + threadIdx.x) >> 5;
  const int CT = (Hout + 15) >> 4;
  const int RT = R >> 4;
  if (wid >= RT * CT) return;
  const int rt = wid / CT, ct = wid % CT;
  const int r0 = rt << 4, c0 = ct << 4;
  const int arow = r0 + (lane & 15);
  const int bcol = c0 + (lane & 15);
  const _Float16* Arow = X + (size_t)arow * Kp;
  const _Float16* Bcol = W + (size_t)bcol * Kp;
  v8f acc = {};
  for (int k0 = 0; k0 < Kp; k0 += 32)
    acc = wmma_acc(load_fragA(Arow + k0, lane), load_fragB(Bcol + k0, lane), acc);
#pragma unroll
  for (int r = 0; r < 8; ++r) {
    int row = r0 + r + ((lane >> 4) << 3);
    int col = c0 + (lane & 15);
    if (col < Hout) {
      float v = fmaxf(acc[r] + bias[col], 0.f);
      if (lens) {
        int t = row % T, n = row / T;
        if (t >= lens[n]) v = 0.f;
      }
      out[(size_t)row * Hout + col] = v;
    }
  }
}

// ---------------------------------------------------------------------------
// Gate precompute: xpre[t,n,0:400] = X(t,n,:) @ Wi^T + bias  (t-major rows)
// mode A (Xn!=null): X row = Xn[(n*T + t)*Kp]   (n-major f16 padded)
// mode B (Yc!=null): X row = Yc[r*Kp]           (t-major f16 padded concat)
// ---------------------------------------------------------------------------
__global__ void k_gates(const _Float16* __restrict__ Xn,
                        const _Float16* __restrict__ Yc,
                        const _Float16* __restrict__ Wi,
                        const float* __restrict__ bias,
                        float* __restrict__ out, int T, int Kp) {
  const int lane = threadIdx.x & 31;
  const int wid = (blockIdx.x * blockDim.x + threadIdx.x) >> 5;
  const int R = T * Nn;
  const int RT = R >> 4;
  if (wid >= RT * 25) return;
  const int rt = wid / 25, ct = wid % 25;
  const int r0 = rt << 4, c0 = ct << 4;
  const int arow = r0 + (lane & 15);
  const int bcol = c0 + (lane & 15);
  const _Float16* Arow;
  if (Xn) {
    int t = arow / Nn, n = arow % Nn;
    Arow = Xn + ((size_t)n * T + t) * Kp;
  } else {
    Arow = Yc + (size_t)arow * Kp;
  }
  const _Float16* Bcol = Wi + (size_t)bcol * Kp;
  v8f acc = {};
  for (int k0 = 0; k0 < Kp; k0 += 32)
    acc = wmma_acc(load_fragA(Arow + k0, lane), load_fragB(Bcol + k0, lane), acc);
#pragma unroll
  for (int r = 0; r < 8; ++r) {
    int row = r0 + r + ((lane >> 4) << 3);
    int col = c0 + (lane & 15);
    out[(size_t)row * H4 + col] = acc[r] + bias[col];
  }
}

// ---------------------------------------------------------------------------
// Persistent LSTM scan. One workgroup (32 waves). h/c in LDS.
// Whht: f16 [col j in 0..399][k in 0..127]  (contiguous k)
// ---------------------------------------------------------------------------
__global__ __launch_bounds__(1024) void k_scan(
    const float* __restrict__ xpre, const _Float16* __restrict__ Whht,
    const float* __restrict__ h0, const float* __restrict__ cinit,
    const int* __restrict__ lens, float* __restrict__ y, int T, int rev) {
  __shared__ __align__(16) _Float16 h_sh[Nn][KPAD];
  __shared__ float c_sh[Nn][Hh];
  __shared__ float z_sh[Nn][H4];
  const int tid = threadIdx.x;
  const int lane = tid & 31;
  const int wid = tid >> 5;

  for (int i = tid; i < Nn * KPAD; i += 1024) {
    int b = i / KPAD, k = i % KPAD;
    h_sh[b][k] = (k < Hh) ? (_Float16)h0[(size_t)b * Hh + k] : (_Float16)0.f;
  }
  for (int i = tid; i < Nn * Hh; i += 1024) c_sh[i / Hh][i % Hh] = cinit[i];
  __syncthreads();

  for (int s = 0; s < T; ++s) {
    const int t = rev ? (T - 1 - s) : s;
    // Phase A: z = h @ Wh^T (2x25 tiles over 32 waves), K padded to 128
    for (int tile = wid; tile < 50; tile += 32) {
      const int rt = tile & 1, ct = tile >> 1;
      const int r0 = rt << 4, cc0 = ct << 4;
      const int arow = r0 + (lane & 15);
      const int bcol = cc0 + (lane & 15);
      const _Float16* Bcol = Whht + (size_t)bcol * KPAD;
      v8f acc = {};
#pragma unroll
      for (int k0 = 0; k0 < KPAD; k0 += 32)
        acc = wmma_acc(load_fragA(&h_sh[arow][k0], lane),
                       load_fragB(Bcol + k0, lane), acc);
#pragma unroll
      for (int r = 0; r < 8; ++r)
        z_sh[r0 + r + ((lane >> 4) << 3)][cc0 + (lane & 15)] = acc[r];
    }
    __syncthreads();
    // Phase B: gates + state update (fp32)
    for (int i = tid; i < Nn * Hh; i += 1024) {
      const int b = i / Hh, hh = i % Hh;
      const float* xp = xpre + ((size_t)t * Nn + b) * H4;
      const float zi = z_sh[b][hh]          + xp[hh];
      const float zf = z_sh[b][Hh + hh]     + xp[Hh + hh];
      const float zg = z_sh[b][2 * Hh + hh] + xp[2 * Hh + hh];
      const float zo = z_sh[b][3 * Hh + hh] + xp[3 * Hh + hh];
      const float si = 1.f / (1.f + __expf(-zi));
      const float sf = 1.f / (1.f + __expf(-zf));
      const float so = 1.f / (1.f + __expf(-zo));
      const float tg = tanhf(zg);
      const float co = c_sh[b][hh];
      const float cn = sf * co + si * tg;
      const float hn = so * tanhf(cn);
      const float m  = (t < lens[b]) ? 1.f : 0.f;
      y[((size_t)t * Nn + b) * Hh + hh] = m * hn;
      const float ho = (float)h_sh[b][hh];
      h_sh[b][hh] = (_Float16)(m * hn + (1.f - m) * ho);
      c_sh[b][hh] = m * cn + (1.f - m) * co;
    }
    __syncthreads();
  }
}

// ---------------------------------------------------------------------------
// Attention scores s[n,l,m] = dot(p_ff[n,l,:], q_ff[n,m,:])
// pffh: (Nn*Ll x 128), qffh: (Nn, 32, 128)
// ---------------------------------------------------------------------------
__global__ void k_sij(const _Float16* __restrict__ pffh,
                      const _Float16* __restrict__ qffh,
                      float* __restrict__ s) {
  const int lane = threadIdx.x & 31;
  const int wid = (blockIdx.x * blockDim.x + threadIdx.x) >> 5;
  if (wid >= Nn * 50) return;               // 25 L-tiles x 2 M-tiles
  const int n = wid / 50, rem = wid % 50;
  const int rt = rem / 2, ct = rem % 2;
  const int r0 = rt << 4, c0 = ct << 4;
  const int arow = r0 + (lane & 15);
  const int bcol = c0 + (lane & 15);
  const _Float16* Arow = pffh + ((size_t)n * Ll + arow) * KPAD;
  const _Float16* Bcol = qffh + ((size_t)n * 32 + bcol) * KPAD;
  v8f acc = {};
#pragma unroll
  for (int k0 = 0; k0 < KPAD; k0 += 32)
    acc = wmma_acc(load_fragA(Arow + k0, lane), load_fragB(Bcol + k0, lane), acc);
#pragma unroll
  for (int r = 0; r < 8; ++r) {
    int row = r0 + r + ((lane >> 4) << 3);
    int col = c0 + (lane & 15);
    if (col < Mm) s[((size_t)n * Ll + row) * Mm + col] = acc[r];
  }
}

// q_i_align[n,l,e] = sum_m a[n,l,m] * q_emd[n,m,e]
// aijh: (Nn, 400, 32), qeTh: (Nn, 304, 32)
__global__ void k_qalign(const _Float16* __restrict__ aijh,
                         const _Float16* __restrict__ qeTh,
                         float* __restrict__ out) {
  const int lane = threadIdx.x & 31;
  const int wid = (blockIdx.x * blockDim.x + threadIdx.x) >> 5;
  if (wid >= Nn * 25 * 19) return;          // 25 L-tiles x 19 E-tiles
  const int n = wid / (25 * 19), rem = wid % (25 * 19);
  const int rt = rem / 19, ct = rem % 19;
  const int r0 = rt << 4, c0 = ct << 4;
  const int arow = r0 + (lane & 15);
  const int bcol = c0 + (lane & 15);
  const _Float16* Arow = aijh + ((size_t)n * Ll + arow) * 32;
  const _Float16* Bcol = qeTh + ((size_t)n * 304 + bcol) * 32;
  v8f acc = wmma_acc(load_fragA(Arow, lane), load_fragB(Bcol, lane), v8f{});
#pragma unroll
  for (int r = 0; r < 8; ++r) {
    int row = r0 + r + ((lane >> 4) << 3);
    int col = c0 + (lane & 15);
    if (col < Ee) out[((size_t)n * Ll + row) * Ee + col] = acc[r];
  }
}

// ---------------------------------------------------------------------------
// Small fp32 kernels
// ---------------------------------------------------------------------------

__global__ void k_softmax(float* __restrict__ x, int rows, int cols) {
  int r = blockIdx.x * blockDim.x + threadIdx.x;
  if (r >= rows) return;
  float* p = x + (size_t)r * cols;
  float mx = p[0];
  for (int c = 1; c < cols; ++c) mx = fmaxf(mx, p[c]);
  float s = 0.f;
  for (int c = 0; c < cols; ++c) { float e = __expf(p[c] - mx); p[c] = e; s += e; }
  float inv = 1.f / s;
  for (int c = 0; c < cols; ++c) p[c] *= inv;
}

__global__ void k_dot(const float* __restrict__ x, const float* __restrict__ w,
                      const float* __restrict__ b, float* __restrict__ out) {
  int idx = blockIdx.x * blockDim.x + threadIdx.x;
  if (idx >= Nn * Mm) return;
  const float* xr = x + (size_t)idx * Hh;
  float s = b[0];
  for (int h = 0; h < Hh; ++h) s += xr[h] * w[h];
  out[idx] = s;
}

__global__ void k_qindep(const float* __restrict__ aj,
                         const float* __restrict__ qdash,
                         float* __restrict__ out) {
  int idx = blockIdx.x * blockDim.x + threadIdx.x;
  if (idx >= Nn * 2 * Hh) return;
  int n = idx / (2 * Hh), k = idx % (2 * Hh);
  float s = 0.f;
  for (int m = 0; m < Mm; ++m)
    s += aj[n * Mm + m] * qdash[((size_t)n * Mm + m) * 2 * Hh + k];
  out[idx] = s;
}

__global__ void k_span(const float* __restrict__ stt,
                       const float* __restrict__ endb,
                       const float* __restrict__ waW,
                       const float* __restrict__ waB,
                       const int* __restrict__ p_lens,
                       float* __restrict__ scores) {
  int idx = blockIdx.x * blockDim.x + threadIdx.x;
  if (idx >= Nn * Ll * Aa) return;
  int a = idx % Aa;
  int l = (idx / Aa) % Ll;
  int n = idx / (Aa * Ll);
  int le = l + a;
  const float* st = stt + ((size_t)n * Ll + l) * Hh;
  float s = waB[0];
  if (le < Ll) {
    const float* en = endb + ((size_t)n * Ll + le) * Hh;
    for (int h = 0; h < Hh; ++h) s += waW[h] * fmaxf(st[h] + en[h], 0.f);
  } else {
    for (int h = 0; h < Hh; ++h) s += waW[h] * fmaxf(st[h], 0.f);
  }
  scores[idx] = (le < p_lens[n]) ? s : 0.f;
}

__global__ void k_logsoftmax(const float* __restrict__ x,
                             float* __restrict__ out, int cols) {
  __shared__ float red[256];
  const int n = blockIdx.x;
  const float* xr = x + (size_t)n * cols;
  float mx = -3.4e38f;
  for (int i = threadIdx.x; i < cols; i += 256) mx = fmaxf(mx, xr[i]);
  red[threadIdx.x] = mx;
  __syncthreads();
  for (int s2 = 128; s2 > 0; s2 >>= 1) {
    if ((int)threadIdx.x < s2)
      red[threadIdx.x] = fmaxf(red[threadIdx.x], red[threadIdx.x + s2]);
    __syncthreads();
  }
  mx = red[0];
  __syncthreads();
  float sm = 0.f;
  for (int i = threadIdx.x; i < cols; i += 256) sm += __expf(xr[i] - mx);
  red[threadIdx.x] = sm;
  __syncthreads();
  for (int s2 = 128; s2 > 0; s2 >>= 1) {
    if ((int)threadIdx.x < s2) red[threadIdx.x] += red[threadIdx.x + s2];
    __syncthreads();
  }
  const float lse = logf(red[0]);
  for (int i = threadIdx.x; i < cols; i += 256)
    out[(size_t)n * cols + i] = xr[i] - mx - lse;
}

// ---------------------------------------------------------------------------
// Host orchestration
// ---------------------------------------------------------------------------

static inline int cdiv(long a, long b) { return (int)((a + b - 1) / b); }

extern "C" void kernel_launch(void* const* d_in, const int* in_sizes, int n_in,
                              void* d_out, int out_size, void* d_ws,
                              size_t ws_size, hipStream_t stream) {
  (void)in_sizes; (void)n_in; (void)out_size; (void)ws_size;
  const float* emd_W = (const float*)d_in[0];
  const float* W_pq  = (const float*)d_in[1];
  const float* b_pq  = (const float*)d_in[2];
  const float* W_qd  = (const float*)d_in[3];
  const float* b_qd  = (const float*)d_in[4];
  const float* wq_W  = (const float*)d_in[5];
  const float* wq_b  = (const float*)d_in[6];
  const float* W_stt = (const float*)d_in[7];
  const float* b_stt = (const float*)d_in[8];
  const float* W_end = (const float*)d_in[9];
  const float* b_end = (const float*)d_in[10];
  const float* wa_W  = (const float*)d_in[11];
  const float* wa_b  = (const float*)d_in[12];
  // 0:q0f 1:q0b 2:q1f 3:q1b 4:p0f 5:p0b 6:p1f 7:p1b
  const float *Wi[8], *Wh[8], *Bb[8];
  for (int i = 0; i < 8; ++i) {
    Wi[i] = (const float*)d_in[13 + 3 * i];
    Wh[i] = (const float*)d_in[14 + 3 * i];
    Bb[i] = (const float*)d_in[15 + 3 * i];
  }
  const float* q_h0 = (const float*)d_in[37];
  const float* q_c0 = (const float*)d_in[38];
  const float* p_h0 = (const float*)d_in[39];
  const float* p_c0 = (const float*)d_in[40];
  const int* p_tok  = (const int*)d_in[41];
  const int* q_tok  = (const int*)d_in[42];
  const int* p_lens = (const int*)d_in[43];
  const int* q_lens = (const int*)d_in[44];

  size_t off = 0;
  auto alloc = [&](size_t bytes) -> char* {
    char* p = (char*)d_ws + off;
    off += (bytes + 255) & ~(size_t)255;
    return p;
  };
  // ---- fp32 scratch ----
  float* qe      = (float*)alloc(sizeof(float) * Nn * Mm * Ee);
  float* qff     = (float*)alloc(sizeof(float) * Nn * Mm * Hh);
  float* xq      = (float*)alloc(sizeof(float) * Mm * Nn * H4);
  float* yq[4];
  for (int i = 0; i < 4; ++i) yq[i] = (float*)alloc(sizeof(float) * Mm * Nn * Hh);
  float* qdash   = (float*)alloc(sizeof(float) * Nn * Mm * 2 * Hh);
  float* qdashff = (float*)alloc(sizeof(float) * Nn * Mm * Hh);
  float* sj      = (float*)alloc(sizeof(float) * Nn * Mm);
  float* qindep  = (float*)alloc(sizeof(float) * Nn * 2 * Hh);
  float* pe      = (float*)alloc(sizeof(float) * (size_t)Nn * Ll * Ee);
  float* pff     = (float*)alloc(sizeof(float) * (size_t)Nn * Ll * Hh);
  float* sij     = (float*)alloc(sizeof(float) * (size_t)Nn * Ll * Mm);
  float* qal     = (float*)alloc(sizeof(float) * (size_t)Nn * Ll * Ee);
  float* xp      = (float*)alloc(sizeof(float) * (size_t)Ll * Nn * H4);
  float* yp[4];
  for (int i = 0; i < 4; ++i)
    yp[i] = (float*)alloc(sizeof(float) * (size_t)Ll * Nn * Hh);
  float* stt     = (float*)alloc(sizeof(float) * (size_t)Nn * Ll * Hh);
  float* endb    = (float*)alloc(sizeof(float) * (size_t)Nn * Ll * Hh);
  float* scores  = (float*)alloc(sizeof(float) * (size_t)Nn * Ll * Aa);
  // ---- f16 padded operands ----
  typedef _Float16 hf;
  hf* W_pqh  = (hf*)alloc(sizeof(hf) * 112 * 320);
  hf* W_qdh  = (hf*)alloc(sizeof(hf) * 112 * 224);
  hf* W_stth = (hf*)alloc(sizeof(hf) * 112 * 224);
  hf* W_endh = (hf*)alloc(sizeof(hf) * 112 * 224);
  const int WiKp[8] = {320, 320, 224, 224, 800, 800, 224, 224};
  hf* WiH[8];
  for (int i = 0; i < 8; ++i) WiH[i] = (hf*)alloc(sizeof(hf) * H4 * WiKp[i]);
  hf* WhT[8];
  for (int i = 0; i < 8; ++i) WhT[i] = (hf*)alloc(sizeof(hf) * H4 * KPAD);
  hf* qeh    = (hf*)alloc(sizeof(hf) * Nn * Mm * 320);
  hf* qeTh   = (hf*)alloc(sizeof(hf) * Nn * 304 * 32);
  hf* qffh   = (hf*)alloc(sizeof(hf) * Nn * 32 * KPAD);
  hf* peh    = (hf*)alloc(sizeof(hf) * (size_t)Nn * Ll * 320);
  hf* pffh   = (hf*)alloc(sizeof(hf) * (size_t)Nn * Ll * KPAD);
  hf* aijh   = (hf*)alloc(sizeof(hf) * (size_t)Nn * Ll * 32);
  hf* qdashh = (hf*)alloc(sizeof(hf) * Nn * Mm * 224);
  hf* psdh   = (hf*)alloc(sizeof(hf) * (size_t)Nn * Ll * 224);
  hf* pstarh = (hf*)alloc(sizeof(hf) * (size_t)Nn * Ll * 800);
  hf* ycatq  = (hf*)alloc(sizeof(hf) * Mm * Nn * 224);
  hf* ycatp  = (hf*)alloc(sizeof(hf) * (size_t)Ll * Nn * 224);

  const int TB = 256;

  // ---- weight conversions ----
  k_h_pad<<<cdiv(112L * 320, TB), TB, 0, stream>>>(W_pq, W_pqh, 1, Hh, 112, Ee, 320);
  k_h_pad<<<cdiv(112L * 224, TB), TB, 0, stream>>>(W_qd, W_qdh, 1, Hh, 112, 2 * Hh, 224);
  k_h_pad<<<cdiv(112L * 224, TB), TB, 0, stream>>>(W_stt, W_stth, 1, Hh, 112, 2 * Hh, 224);
  k_h_pad<<<cdiv(112L * 224, TB), TB, 0, stream>>>(W_end, W_endh, 1, Hh, 112, 2 * Hh, 224);
  const int WiKs[8] = {Ee, Ee, 2 * Hh, 2 * Hh, 800, 800, 2 * Hh, 2 * Hh};
  for (int i = 0; i < 8; ++i) {
    k_h_pad<<<cdiv((long)H4 * WiKp[i], TB), TB, 0, stream>>>(
        Wi[i], WiH[i], 1, H4, H4, WiKs[i], WiKp[i]);
    k_wh2half<<<cdiv((long)H4 * KPAD, TB), TB, 0, stream>>>(Wh[i], WhT[i]);
  }

  // ---- question path ----
  k_gather<<<cdiv((long)Nn * Mm * Ee, TB), TB, 0, stream>>>(emd_W, q_tok, qe, Nn * Mm);
  k_h_pad<<<cdiv((long)Nn * Mm * 320, TB), TB, 0, stream>>>(qe, qeh, 1, Nn * Mm, Nn * Mm, Ee, 320);
  k_qeT<<<cdiv((long)Nn * 304 * 32, TB), TB, 0, stream>>>(qe, qeTh);
  {
    long waves = (long)((Nn * Mm) / 16) * 7;
    k_ff<<<cdiv(waves * 32, TB), TB, 0, stream>>>(qeh, W_pqh, b_pq, q_lens, qff,
                                                  Nn * Mm, 320, Hh, Mm);
  }
  k_h_pad<<<cdiv((long)Nn * 32 * KPAD, TB), TB, 0, stream>>>(qff, qffh, Nn, Mm, 32, Hh, KPAD);
  {
    long waves = (long)((Mm * Nn) / 16) * 25;
    k_gates<<<cdiv(waves * 32, TB), TB, 0, stream>>>(qeh, nullptr, WiH[0], Bb[0], xq, Mm, 320);
    k_scan<<<1, 1024, 0, stream>>>(xq, WhT[0], q_h0 + 0 * Nn * Hh, q_c0 + 0 * Nn * Hh, q_lens, yq[0], Mm, 0);
    k_gates<<<cdiv(waves * 32, TB), TB, 0, stream>>>(qeh, nullptr, WiH[1], Bb[1], xq, Mm, 320);
    k_scan<<<1, 1024, 0, stream>>>(xq, WhT[1], q_h0 + 1 * Nn * Hh, q_c0 + 1 * Nn * Hh, q_lens, yq[1], Mm, 1);
    k_concat_h_t<<<cdiv((long)Mm * Nn * 224, TB), TB, 0, stream>>>(yq[0], yq[1], ycatq, Mm);
    k_gates<<<cdiv(waves * 32, TB), TB, 0, stream>>>(nullptr, ycatq, WiH[2], Bb[2], xq, Mm, 224);
    k_scan<<<1, 1024, 0, stream>>>(xq, WhT[2], q_h0 + 2 * Nn * Hh, q_c0 + 2 * Nn * Hh, q_lens, yq[2], Mm, 0);
    k_gates<<<cdiv(waves * 32, TB), TB, 0, stream>>>(nullptr, ycatq, WiH[3], Bb[3], xq, Mm, 224);
    k_scan<<<1, 1024, 0, stream>>>(xq, WhT[3], q_h0 + 3 * Nn * Hh, q_c0 + 3 * Nn * Hh, q_lens, yq[3], Mm, 1);
  }
  k_concat<<<cdiv((long)Nn * Mm * 2 * Hh, TB), TB, 0, stream>>>(yq[2], yq[3], qdash, Mm);
  k_concat_h_n<<<cdiv((long)Nn * Mm * 224, TB), TB, 0, stream>>>(yq[2], yq[3], qdashh, Mm);
  {
    long waves = (long)((Nn * Mm) / 16) * 7;
    k_ff<<<cdiv(waves * 32, TB), TB, 0, stream>>>(qdashh, W_qdh, b_qd, q_lens, qdashff,
                                                  Nn * Mm, 224, Hh, Mm);
  }
  k_dot<<<cdiv((long)Nn * Mm, TB), TB, 0, stream>>>(qdashff, wq_W, wq_b, sj);
  k_softmax<<<cdiv((long)Nn, 64), 64, 0, stream>>>(sj, Nn, Mm);
  k_qindep<<<cdiv((long)Nn * 2 * Hh, TB), TB, 0, stream>>>(sj, qdash, qindep);

  // ---- passage path + attention ----
  k_gather<<<cdiv((long)Nn * Ll * Ee, TB), TB, 0, stream>>>(emd_W, p_tok, pe, Nn * Ll);
  k_h_pad<<<cdiv((long)Nn * Ll * 320, TB), TB, 0, stream>>>(pe, peh, 1, Nn * Ll, Nn * Ll, Ee, 320);
  {
    long waves = (long)((Nn * Ll) / 16) * 7;
    k_ff<<<cdiv(waves * 32, TB), TB, 0, stream>>>(peh, W_pqh, b_pq, p_lens, pff,
                                                  Nn * Ll, 320, Hh, Ll);
  }
  k_h_pad<<<cdiv((long)Nn * Ll * KPAD, TB), TB, 0, stream>>>(pff, pffh, 1, Nn * Ll, Nn * Ll, Hh, KPAD);
  k_sij<<<cdiv((long)Nn * 50 * 32, TB), TB, 0, stream>>>(pffh, qffh, sij);
  k_softmax<<<cdiv((long)Nn * Ll, 64), 64, 0, stream>>>(sij, Nn * Ll, Mm);
  k_h_pad<<<cdiv((long)Nn * Ll * 32, TB), TB, 0, stream>>>(sij, aijh, Nn, Ll, Ll, Mm, 32);
  k_qalign<<<cdiv((long)Nn * 25 * 19 * 32, TB), TB, 0, stream>>>(aijh, qeTh, qal);
  k_pstar_h<<<cdiv((long)Nn * Ll * 800, TB), TB, 0, stream>>>(pe, qal, qindep, pstarh);

  // ---- passage BiLSTM ----
  {
    long waves = (long)((Ll * Nn) / 16) * 25;
    k_gates<<<cdiv(waves * 32, TB), TB, 0, stream>>>(pstarh, nullptr, WiH[4], Bb[4], xp, Ll, 800);
    k_scan<<<1, 1024, 0, stream>>>(xp, WhT[4], p_h0 + 0 * Nn * Hh, p_c0 + 0 * Nn * Hh, p_lens, yp[0], Ll, 0);
    k_gates<<<cdiv(waves * 32, TB), TB, 0, stream>>>(pstarh, nullptr, WiH[5], Bb[5], xp, Ll, 800);
    k_scan<<<1, 1024, 0, stream>>>(xp, WhT[5], p_h0 + 1 * Nn * Hh, p_c0 + 1 * Nn * Hh, p_lens, yp[1], Ll, 1);
    k_concat_h_t<<<cdiv((long)Ll * Nn * 224, TB), TB, 0, stream>>>(yp[0], yp[1], ycatp, Ll);
    k_gates<<<cdiv(waves * 32, TB), TB, 0, stream>>>(nullptr, ycatp, WiH[6], Bb[6], xp, Ll, 224);
    k_scan<<<1, 1024, 0, stream>>>(xp, WhT[6], p_h0 + 2 * Nn * Hh, p_c0 + 2 * Nn * Hh, p_lens, yp[2], Ll, 0);
    k_gates<<<cdiv(waves * 32, TB), TB, 0, stream>>>(nullptr, ycatp, WiH[7], Bb[7], xp, Ll, 224);
    k_scan<<<1, 1024, 0, stream>>>(xp, WhT[7], p_h0 + 3 * Nn * Hh, p_c0 + 3 * Nn * Hh, p_lens, yp[3], Ll, 1);
  }
  k_concat_h_n<<<cdiv((long)Nn * Ll * 224, TB), TB, 0, stream>>>(yp[2], yp[3], psdh, Ll);

  // ---- span scorer ----
  {
    long waves = (long)((Nn * Ll) / 16) * 7;
    k_ff<<<cdiv(waves * 32, TB), TB, 0, stream>>>(psdh, W_stth, b_stt, nullptr, stt,
                                                  Nn * Ll, 224, Hh, Ll);
    k_ff<<<cdiv(waves * 32, TB), TB, 0, stream>>>(psdh, W_endh, b_end, nullptr, endb,
                                                  Nn * Ll, 224, Hh, Ll);
  }
  k_span<<<cdiv((long)Nn * Ll * Aa, TB), TB, 0, stream>>>(stt, endb, wa_W, wa_b,
                                                          p_lens, scores);
  k_logsoftmax<<<Nn, 256, 0, stream>>>(scores, (float*)d_out, Ll * Aa);
}
